// VATGNNExpert_20538533609919
// MI455X (gfx1250) — compile-verified, matrix operand order checked by reading, not run on hardware
//
#include <hip/hip_runtime.h>

typedef __attribute__((ext_vector_type(16))) _Float16 v16h;
typedef __attribute__((ext_vector_type(8)))  _Float16 v8h;
typedef __attribute__((ext_vector_type(8)))  float    v8f;

#define GNN_N      100000
#define GNN_E      800000
#define GNN_HID    128
#define GNN_HEADS  8
#define GNN_D      16
#define GNN_LAYERS 5
#define NEG_SLOPE  0.2f

// ---------------------------------------------------------------- helpers
__device__ __forceinline__ float gelu_tanh(float x) {
    // jax.nn.gelu default (approximate=True)
    float x3 = x * x * x;
    return 0.5f * x * (1.0f + tanhf(0.7978845608028654f * (x + 0.044715f * x3)));
}

// order-preserving float -> uint key (for atomic segment-max)
__device__ __forceinline__ unsigned fkey(float f) {
    unsigned u = __float_as_uint(f);
    unsigned mask = (u & 0x80000000u) ? 0xFFFFFFFFu : 0x80000000u;
    return u ^ mask;
}
__device__ __forceinline__ float fdecode(unsigned k) {
    unsigned u = (k & 0x80000000u) ? (k ^ 0x80000000u) : ~k;
    return __uint_as_float(u);
}

// ---------------------------------------------------------------- WMMA GEMM
// C[M,128] = act(A[M,128] @ W[128,128] + bias), f16 WMMA, f32 accumulate.
// Workgroup = 256 threads = 8 waves; each wave owns 16 rows x 128 cols
// (8 N-tiles x 4 K-chunks = 32 v_wmma_f32_16x16x32_f16 per wave).
__global__ __launch_bounds__(256) void k_gemm128_wmma(
    const float* __restrict__ A, const float* __restrict__ W,
    const float* __restrict__ bias, float* __restrict__ C,
    int M, int act)
{
    // W transposed into LDS as f16: Wt[n][k], padded stride 144 halves
    __shared__ _Float16 Wt[128 * 144];
    const int tid = threadIdx.x;
    #pragma unroll 8
    for (int i = tid; i < 128 * 128; i += 256) {
        int k = i >> 7, n = i & 127;
        Wt[n * 144 + k] = (_Float16)W[i];
    }
    __syncthreads();

    const int wave  = tid >> 5;
    const int lane  = tid & 31;
    const int mrow  = lane & 15;
    const int khalf = (lane >> 4) << 3;          // 0 or 8 (A-fragment K sub-block)
    const long long row = (long long)blockIdx.x * 128 + wave * 16 + mrow;
    const bool rowOK = row < M;
    const float* Arow = A + row * 128;

    v8f zero;
    #pragma unroll
    for (int r = 0; r < 8; ++r) zero[r] = 0.0f;
    v8f acc[8];
    #pragma unroll
    for (int nt = 0; nt < 8; ++nt) acc[nt] = zero;

    #pragma unroll
    for (int kc = 0; kc < 4; ++kc) {
        const int k0 = kc * 32 + khalf;
        v16h a;
        if (rowOK) {
            float4 f0 = *(const float4*)(Arow + k0);
            float4 f1 = *(const float4*)(Arow + k0 + 4);
            float4 f2 = *(const float4*)(Arow + k0 + 16);
            float4 f3 = *(const float4*)(Arow + k0 + 20);
            a[0]=(_Float16)f0.x; a[1]=(_Float16)f0.y; a[2]=(_Float16)f0.z; a[3]=(_Float16)f0.w;
            a[4]=(_Float16)f1.x; a[5]=(_Float16)f1.y; a[6]=(_Float16)f1.z; a[7]=(_Float16)f1.w;
            a[8]=(_Float16)f2.x; a[9]=(_Float16)f2.y; a[10]=(_Float16)f2.z; a[11]=(_Float16)f2.w;
            a[12]=(_Float16)f3.x; a[13]=(_Float16)f3.y; a[14]=(_Float16)f3.z; a[15]=(_Float16)f3.w;
        } else {
            #pragma unroll
            for (int j = 0; j < 16; ++j) a[j] = (_Float16)0.0f;
        }
        #pragma unroll
        for (int nt = 0; nt < 8; ++nt) {
            const _Float16* bp = &Wt[(nt * 16 + mrow) * 144 + k0];
            v8h blo = *(const v8h*)bp;        // 16B-aligned ds loads
            v8h bhi = *(const v8h*)(bp + 16);
            v16h b;
            #pragma unroll
            for (int j = 0; j < 8; ++j) { b[j] = blo[j]; b[j + 8] = bhi[j]; }
            acc[nt] = __builtin_amdgcn_wmma_f32_16x16x32_f16(
                false, a, false, b, (short)0, acc[nt], false, false);
        }
    }

    // C/D layout: VGPR r -> (M=r, N=lane) lanes 0-15 ; (M=r+8, N=lane-16) lanes 16-31
    const long long srow0 = (long long)blockIdx.x * 128 + wave * 16 + ((lane >> 4) << 3);
    const int col0 = lane & 15;
    #pragma unroll
    for (int nt = 0; nt < 8; ++nt) {
        const int col = nt * 16 + col0;
        const float bv = bias[col];
        #pragma unroll
        for (int r = 0; r < 8; ++r) {
            long long rr = srow0 + r;
            if (rr < M) {
                float v = acc[nt][r] + bv;
                if (act == 1)      v = gelu_tanh(v);
                else if (act == 2) v = tanhf(v);
                C[rr * 128 + col] = v;
            }
        }
    }
}

// ---------------------------------------------------------------- LayerNorm
// one wave32 per 128-wide row, 4 elements per lane, shuffle-tree reduce
__global__ __launch_bounds__(256) void k_layernorm128(
    const float* __restrict__ x, const float* __restrict__ g,
    const float* __restrict__ b, float* __restrict__ y, int M)
{
    const int wave = threadIdx.x >> 5, lane = threadIdx.x & 31;
    const long long row = (long long)blockIdx.x * 8 + wave;
    if (row >= M) return;
    float4 v = ((const float4*)(x + row * 128))[lane];
    float s  = v.x + v.y + v.z + v.w;
    float s2 = v.x*v.x + v.y*v.y + v.z*v.z + v.w*v.w;
    #pragma unroll
    for (int off = 16; off > 0; off >>= 1) {
        s  += __shfl_xor(s,  off, 32);
        s2 += __shfl_xor(s2, off, 32);
    }
    const float mean = s * (1.0f / 128.0f);
    const float var  = s2 * (1.0f / 128.0f) - mean * mean;
    const float inv  = rsqrtf(var + 1e-5f);
    float4 gg = ((const float4*)g)[lane];
    float4 bb = ((const float4*)b)[lane];
    float4 o;
    o.x = (v.x - mean) * inv * gg.x + bb.x;
    o.y = (v.y - mean) * inv * gg.y + bb.y;
    o.z = (v.z - mean) * inv * gg.z + bb.z;
    o.w = (v.w - mean) * inv * gg.w + bb.w;
    ((float4*)(y + row * 128))[lane] = o;
}

// ---------------------------------------------------------------- edge pass 1
// logits[e,h] = sum_d att[h,d] * leaky_relu(xl[src,h,d] + xr[dst,h,d])
// + atomic segment-max into mkey[dst,h]
__global__ __launch_bounds__(256) void k_edge_logits(
    const float* __restrict__ xl, const float* __restrict__ xr,
    const float* __restrict__ att, const long long* __restrict__ ei,
    float* __restrict__ logits, unsigned* __restrict__ mkey)
{
    const int gid = blockIdx.x * 256 + threadIdx.x;
    const int tot = (GNN_E + GNN_N) * GNN_HEADS;
    if (gid >= tot) return;
    const int e = gid >> 3, h = gid & 7;
    long long s, d;
    if (e < GNN_E) { s = ei[e]; d = ei[GNN_E + e]; } else { s = d = e - GNN_E; }
    const float* pl = xl + s * GNN_HID + h * GNN_D;
    const float* pr = xr + d * GNN_HID + h * GNN_D;
    const float* pa = att + h * GNN_D;
    float acc = 0.0f;
    #pragma unroll
    for (int i = 0; i < GNN_D; ++i) {
        float v = pl[i] + pr[i];
        v = v > 0.0f ? v : NEG_SLOPE * v;
        acc += pa[i] * v;
    }
    logits[gid] = acc;
    atomicMax(&mkey[(int)d * GNN_HEADS + h], fkey(acc));
}

// ---------------------------------------------------------------- edge pass 2
// a = exp(logit - max[dst,h]); segment-sum into ssum[dst,h]
__global__ __launch_bounds__(256) void k_edge_exp(
    float* __restrict__ logits, const unsigned* __restrict__ mkey,
    float* __restrict__ ssum, const long long* __restrict__ ei)
{
    const int gid = blockIdx.x * 256 + threadIdx.x;
    const int tot = (GNN_E + GNN_N) * GNN_HEADS;
    if (gid >= tot) return;
    const int e = gid >> 3, h = gid & 7;
    long long d = (e < GNN_E) ? ei[GNN_E + e] : (long long)(e - GNN_E);
    const float m = fdecode(mkey[(int)d * GNN_HEADS + h]);
    const float a = expf(logits[gid] - m);
    logits[gid] = a;
    atomicAdd(&ssum[(int)d * GNN_HEADS + h], a);
}

// ---------------------------------------------------------------- edge pass 3
// agg[dst,h,:] += xl[src,h,:] * (a / (ssum[dst,h] + eps))
__global__ __launch_bounds__(256) void k_edge_accum(
    const float* __restrict__ xl, const float* __restrict__ logits,
    const float* __restrict__ ssum, const long long* __restrict__ ei,
    float* __restrict__ agg)
{
    const int gid = blockIdx.x * 256 + threadIdx.x;
    const int tot = (GNN_E + GNN_N) * GNN_HEADS;
    if (gid >= tot) return;
    const int e = gid >> 3, h = gid & 7;
    long long s, d;
    if (e < GNN_E) { s = ei[e]; d = ei[GNN_E + e]; } else { s = d = e - GNN_E; }
    const float alpha = logits[gid] / (ssum[(int)d * GNN_HEADS + h] + 1e-16f);
    const float* pl = xl + s * GNN_HID + h * GNN_D;
    float*       po = agg + d * GNN_HID + h * GNN_D;
    #pragma unroll
    for (int i = 0; i < GNN_D; ++i) atomicAdd(&po[i], pl[i] * alpha);
}

// ---------------------------------------------------------------- residual
__global__ __launch_bounds__(256) void k_residual(
    float* __restrict__ h, const float* __restrict__ agg,
    const float* __restrict__ cb)
{
    const long long i = (long long)blockIdx.x * 256 + threadIdx.x;
    if (i >= (long long)GNN_N * GNN_HID) return;
    h[i] += agg[i] + cb[(int)(i & 127)];
}

// ---------------------------------------------------------------- launcher
extern "C" void kernel_launch(void* const* d_in, const int* in_sizes, int n_in,
                              void* d_out, int out_size, void* d_ws, size_t ws_size,
                              hipStream_t stream)
{
    (void)in_sizes; (void)n_in; (void)out_size; (void)ws_size;
    const float*     x     = (const float*)d_in[0];
    const long long* ei    = (const long long*)d_in[1];   // int64 edge_index [2,E]
    const float*     W_in  = (const float*)d_in[2];
    const float*     b_in  = (const float*)d_in[3];
    const float*     g_in  = (const float*)d_in[4];
    const float*     be_in = (const float*)d_in[5];
    const float*     ln_g  = (const float*)d_in[6];
    const float*     ln_b  = (const float*)d_in[7];
    const float*     Wl    = (const float*)d_in[8];
    const float*     bl    = (const float*)d_in[9];
    const float*     Wr    = (const float*)d_in[10];
    const float*     br    = (const float*)d_in[11];
    const float*     att   = (const float*)d_in[12];
    const float*     convb = (const float*)d_in[13];
    const float*     W_sq  = (const float*)d_in[14];
    const float*     b_sq  = (const float*)d_in[15];
    const float*     g_f   = (const float*)d_in[16];
    const float*     b_f   = (const float*)d_in[17];

    const int N = GNN_N, E = GNN_E;
    const long long NH = (long long)N * GNN_HID;

    // workspace carve (~291 MB)
    char* p = (char*)d_ws;
    float*    h    = (float*)p;    p += NH * 4;
    float*    hn   = (float*)p;    p += NH * 4;
    float*    xl   = (float*)p;    p += NH * 4;
    float*    xr   = (float*)p;    p += NH * 4;
    float*    agg  = (float*)p;    p += NH * 4;
    float*    lgt  = (float*)p;    p += (long long)(E + N) * GNN_HEADS * 4;
    unsigned* mkey = (unsigned*)p; p += (long long)N * GNN_HEADS * 4;
    float*    ssum = (float*)p;    p += (long long)N * GNN_HEADS * 4;

    const dim3 blk(256);
    const int gGemm = (N + 127) / 128;
    const int gLN   = (N + 7) / 8;
    const int gE    = ((E + N) * GNN_HEADS + 255) / 256;
    const int gR    = (int)((NH + 255) / 256);

    // input projection: GELU(x @ W_in + b_in) then LayerNorm -> h
    k_gemm128_wmma<<<gGemm, blk, 0, stream>>>(x, W_in, b_in, hn, N, /*act=GELU*/1);
    k_layernorm128<<<gLN, blk, 0, stream>>>(hn, g_in, be_in, h, N);

    for (int l = 0; l < GNN_LAYERS; ++l) {
        k_layernorm128<<<gLN, blk, 0, stream>>>(h, ln_g + l * 128, ln_b + l * 128, hn, N);
        k_gemm128_wmma<<<gGemm, blk, 0, stream>>>(hn, Wl + (long long)l * 128 * 128,
                                                  bl + l * 128, xl, N, 0);
        k_gemm128_wmma<<<gGemm, blk, 0, stream>>>(hn, Wr + (long long)l * 128 * 128,
                                                  br + l * 128, xr, N, 0);
        hipMemsetAsync(mkey, 0, (size_t)N * GNN_HEADS * 4, stream);
        hipMemsetAsync(ssum, 0, (size_t)N * GNN_HEADS * 4, stream);
        hipMemsetAsync(agg,  0, (size_t)NH * 4, stream);
        k_edge_logits<<<gE, blk, 0, stream>>>(xl, xr, att + l * GNN_HEADS * GNN_D, ei, lgt, mkey);
        k_edge_exp   <<<gE, blk, 0, stream>>>(lgt, mkey, ssum, ei);
        k_edge_accum <<<gE, blk, 0, stream>>>(xl, lgt, ssum, ei, agg);
        k_residual   <<<gR, blk, 0, stream>>>(h, agg, convb + l * 128);
    }

    // squeezer: tanh(h @ W_sq + b_sq) then final LayerNorm -> d_out
    k_gemm128_wmma<<<gGemm, blk, 0, stream>>>(h, W_sq, b_sq, hn, N, /*act=tanh*/2);
    k_layernorm128<<<gLN, blk, 0, stream>>>(hn, g_f, b_f, (float*)d_out, N);
}
